// DeepGraphSAGE_62251255988404
// MI455X (gfx1250) — matrix operand order repaired and checked by reading, not compiled, not run on hardware
//
#include <hip/hip_runtime.h>

typedef __attribute__((ext_vector_type(16))) _Float16 v16h;
typedef __attribute__((ext_vector_type(8)))  _Float16 v8h;
typedef __attribute__((ext_vector_type(4)))  _Float16 v4h;
typedef __attribute__((ext_vector_type(8)))  float    v8f;

#define D_HID 128
#define D_OUT 64
#define BN_EPS 1e-5f
#define LDS_LDH 136   // padded row pitch in f16 elements (272B = 16B-aligned, de-conflicts banks)

// ---------------------------------------------------------------------------
// Utility: zero a float buffer
// ---------------------------------------------------------------------------
__global__ void zero_f32_kernel(float* __restrict__ p, int n) {
    int i = blockIdx.x * blockDim.x + threadIdx.x;
    if (i < n) p[i] = 0.0f;
}

// ---------------------------------------------------------------------------
// In-degree via atomics (edge_index passed as int32 per harness convention)
// ---------------------------------------------------------------------------
__global__ void degree_kernel(const int* __restrict__ ei, float* __restrict__ deg, int E) {
    int e = blockIdx.x * blockDim.x + threadIdx.x;
    if (e < E) {
        int dst = ei[(size_t)E + e];
        atomicAdd(&deg[dst], 1.0f);
    }
}

__global__ void invdeg_kernel(float* __restrict__ deg, int N) {
    int i = blockIdx.x * blockDim.x + threadIdx.x;
    if (i < N) deg[i] = 1.0f / fmaxf(deg[i], 1.0f);
}

// ---------------------------------------------------------------------------
// Pack an f32 [K x Ncols] row-major weight matrix into the WMMA B-operand
// register layout (f16). Tile = 32(K) x 16(N). Per lane (=column, split at
// lane 16 for K+8) 16 contiguous f16 so the GEMM does one 32B load per tile.
// ---------------------------------------------------------------------------
__global__ void pack_b16_kernel(const float* __restrict__ W, _Float16* __restrict__ out,
                                int Kt, int Nt) {
    int idx = blockIdx.x * blockDim.x + threadIdx.x;
    int total = Kt * Nt * 32;
    if (idx >= total) return;
    int lane = idx & 31;
    int tile = idx >> 5;
    int nt = tile % Nt;
    int kt = tile / Nt;
    int kb = (lane >> 4) * 8;
    int ncols = Nt * 16;
    const float* w = W + (size_t)(kt * 32 + kb) * ncols + nt * 16 + (lane & 15);
    _Float16* o = out + ((size_t)tile * 32 + lane) * 16;
#pragma unroll
    for (int i = 0; i < 16; ++i) o[i] = (_Float16)w[(size_t)(i + (i >> 3) * 8) * ncols];
}

// ---------------------------------------------------------------------------
// Edge scatter: agg[dst] += h[src]  (mean scaling folded into GEMM A-load).
// One thread per (edge, 4-float chunk): coalesced float4 gather, 4 f32 atomics
// that land in the 192MB L2 (working set fits). Edge-index stream is
// sequential -> far-ahead prefetch into GL2 (global_prefetch_b8).
// ---------------------------------------------------------------------------
__global__ void scatter_kernel(const float* __restrict__ hin, const int* __restrict__ ei,
                               float* __restrict__ agg, int E) {
    int t = blockIdx.x * blockDim.x + threadIdx.x;
    int e = t >> 5;
    if (e >= E) return;
    int c = (t & 31) << 2;
    if ((t & 31) == 0 && e + 2048 < E) {
        __builtin_prefetch(ei + e + 2048, 0, 1);
        __builtin_prefetch(ei + (size_t)E + e + 2048, 0, 1);
    }
    int src = ei[e];
    int dst = ei[(size_t)E + e];
    float4 v = *(const float4*)(hin + (size_t)src * D_HID + c);
    float* p = agg + (size_t)dst * D_HID + c;
    atomicAdd(p + 0, v.x);
    atomicAdd(p + 1, v.y);
    atomicAdd(p + 2, v.z);
    atomicAdd(p + 3, v.w);
}

// ---------------------------------------------------------------------------
// Fused SAGE layer GEMM:  hout = (agg*inv_deg) @ Wl + b + hin @ Wr
// Block = 256 threads = 8 waves; block owns 16 rows, wave w owns columns
// [16w,16w+16). A operands are converted to f16 ONCE during cooperative
// staging (8 waves reuse them -> 8x less cvt work than converting per wave)
// and read back as two 16B ds_loads per fragment. Two independent WMMA
// accumulator chains (agg@Wl, h@Wr) to hide WMMA latency / hazard NOPs.
// Emits BN partial sums via shfl_xor(16) + L2 atomics.
// ---------------------------------------------------------------------------
__global__ __launch_bounds__(256) void sage_gemm_kernel(
    const float* __restrict__ agg, const float* __restrict__ hin,
    const float* __restrict__ inv_deg,
    const _Float16* __restrict__ Wl, const _Float16* __restrict__ Wr,
    const float* __restrict__ bias,
    float* __restrict__ hout,
    float* __restrict__ colsum, float* __restrict__ colsumsq,
    int N)
{
    __shared__ _Float16 ldsA[16 * LDS_LDH];
    __shared__ _Float16 ldsH[16 * LDS_LDH];

    int m0 = blockIdx.x * 16;
    int tid = threadIdx.x;

    // cooperative stage + f32->f16 convert: 16 rows x 32 float4 per matrix
    for (int i = tid; i < 16 * 32; i += 256) {
        int r = i >> 5, c4 = (i & 31) << 2;
        int row = m0 + r;
        if (row >= N) row = N - 1;              // clamp (N % 16 == 0 in practice)
        float s = inv_deg[row];
        float4 a = *(const float4*)(agg + (size_t)row * D_HID + c4);
        float4 h = *(const float4*)(hin + (size_t)row * D_HID + c4);
        v4h a4, h4;
        a4[0] = (_Float16)(a.x * s); a4[1] = (_Float16)(a.y * s);
        a4[2] = (_Float16)(a.z * s); a4[3] = (_Float16)(a.w * s);
        h4[0] = (_Float16)h.x; h4[1] = (_Float16)h.y;
        h4[2] = (_Float16)h.z; h4[3] = (_Float16)h.w;
        *(v4h*)(ldsA + r * LDS_LDH + c4) = a4;
        *(v4h*)(ldsH + r * LDS_LDH + c4) = h4;
    }
    __syncthreads();

    int wave = tid >> 5;
    int lane = tid & 31;
    int n0 = wave * 16;
    int m  = lane & 15;             // A row within tile
    int kb = (lane >> 4) * 8;       // A K sub-offset (lanes 16-31 -> +8)
    int col = lane & 15;            // C/D column within tile

    v8f c0 = {};                    // agg @ Wl chain
    v8f c1 = {};                    // h   @ Wr chain
#pragma unroll
    for (int kt = 0; kt < 4; ++kt) {
        const _Float16* pa = ldsA + m * LDS_LDH + kt * 32 + kb;
        const _Float16* ph = ldsH + m * LDS_LDH + kt * 32 + kb;
        v8h alo = *(const v8h*)(pa);
        v8h ahi = *(const v8h*)(pa + 16);
        v8h hlo = *(const v8h*)(ph);
        v8h hhi = *(const v8h*)(ph + 16);
        v16h a = __builtin_shufflevector(alo, ahi, 0,1,2,3,4,5,6,7,8,9,10,11,12,13,14,15);
        v16h h = __builtin_shufflevector(hlo, hhi, 0,1,2,3,4,5,6,7,8,9,10,11,12,13,14,15);
        v16h bl = *(const v16h*)(Wl + (((size_t)kt * 8 + wave) * 32 + lane) * 16);
        v16h br = *(const v16h*)(Wr + (((size_t)kt * 8 + wave) * 32 + lane) * 16);
        c0 = __builtin_amdgcn_wmma_f32_16x16x32_f16(false, a, false, bl, (short)0, c0, false, false);
        c1 = __builtin_amdgcn_wmma_f32_16x16x32_f16(false, h, false, br, (short)0, c1, false, false);
    }

    float b = bias[n0 + col];
    float lsum = 0.0f, lsq = 0.0f;
#pragma unroll
    for (int r = 0; r < 8; ++r) {
        int row = m0 + r + ((lane >> 4) << 3);
        float v = c0[r] + c1[r] + b;
        if (row < N) {
            hout[(size_t)row * D_HID + n0 + col] = v;
            lsum += v;
            lsq  += v * v;
        }
    }
    // lanes l and l^16 hold the same column (different row halves)
    lsum += __shfl_xor(lsum, 16, 32);
    lsq  += __shfl_xor(lsq, 16, 32);
    if (lane < 16) {
        atomicAdd(&colsum[n0 + col], lsum);
        atomicAdd(&colsumsq[n0 + col], lsq);
    }
}

// ---------------------------------------------------------------------------
// BatchNorm (batch statistics) + ReLU, in place. var = E[x^2] - mean^2.
// ---------------------------------------------------------------------------
__global__ void bn_relu_kernel(float* __restrict__ h,
                               const float* __restrict__ colsum,
                               const float* __restrict__ colsumsq,
                               const float* __restrict__ gamma,
                               const float* __restrict__ beta,
                               int N)
{
    int idx = blockIdx.x * blockDim.x + threadIdx.x;   // over N * 32 float4
    int total = N * (D_HID / 4);
    if (idx >= total) return;
    int c4 = (idx & 31) << 2;
    float invN = 1.0f / (float)N;
    float4 v = *(float4*)(h + (size_t)idx * 4);
    float* vp = &v.x;
#pragma unroll
    for (int j = 0; j < 4; ++j) {
        int cch = c4 + j;
        float mean = colsum[cch] * invN;
        float var  = colsumsq[cch] * invN - mean * mean;
        float t = (vp[j] - mean) * rsqrtf(var + BN_EPS) * gamma[cch] + beta[cch];
        vp[j] = fmaxf(t, 0.0f);
    }
    *(float4*)(h + (size_t)idx * 4) = v;
}

// ---------------------------------------------------------------------------
// Final projection: out = h @ W_o + b_o   ([N,128] x [128,64])
// Block = 128 threads = 4 waves (4 column tiles of 16). f16-staged A in LDS.
// ---------------------------------------------------------------------------
__global__ __launch_bounds__(128) void out_gemm_kernel(
    const float* __restrict__ hin, const _Float16* __restrict__ Wo,
    const float* __restrict__ bo, float* __restrict__ out, int N)
{
    __shared__ _Float16 ldsH[16 * LDS_LDH];
    int m0 = blockIdx.x * 16;
    int tid = threadIdx.x;

    for (int i = tid; i < 16 * 32; i += 128) {
        int r = i >> 5, c4 = (i & 31) << 2;
        int row = m0 + r;
        if (row >= N) row = N - 1;
        float4 h = *(const float4*)(hin + (size_t)row * D_HID + c4);
        v4h h4;
        h4[0] = (_Float16)h.x; h4[1] = (_Float16)h.y;
        h4[2] = (_Float16)h.z; h4[3] = (_Float16)h.w;
        *(v4h*)(ldsH + r * LDS_LDH + c4) = h4;
    }
    __syncthreads();

    int wave = tid >> 5;
    int lane = tid & 31;
    int n0 = wave * 16;
    int m  = lane & 15;
    int kb = (lane >> 4) * 8;
    int col = lane & 15;

    v8f c = {};
#pragma unroll
    for (int kt = 0; kt < 4; ++kt) {
        const _Float16* ph = ldsH + m * LDS_LDH + kt * 32 + kb;
        v8h lo = *(const v8h*)(ph);
        v8h hi = *(const v8h*)(ph + 16);
        v16h a = __builtin_shufflevector(lo, hi, 0,1,2,3,4,5,6,7,8,9,10,11,12,13,14,15);
        v16h b = *(const v16h*)(Wo + (((size_t)kt * 4 + wave) * 32 + lane) * 16);
        c = __builtin_amdgcn_wmma_f32_16x16x32_f16(false, a, false, b, (short)0, c, false, false);
    }

    float b = bo[n0 + col];
#pragma unroll
    for (int r = 0; r < 8; ++r) {
        int row = m0 + r + ((lane >> 4) << 3);
        if (row < N) out[(size_t)row * D_OUT + n0 + col] = c[r] + b;
    }
}

// ---------------------------------------------------------------------------
// Host orchestration
// ---------------------------------------------------------------------------
extern "C" void kernel_launch(void* const* d_in, const int* in_sizes, int n_in,
                              void* d_out, int out_size, void* d_ws, size_t ws_size,
                              hipStream_t stream) {
    const float* x     = (const float*)d_in[0];
    const int*   ei    = (const int*)d_in[1];     // edge_index [2,E]
    const float* W_l   = (const float*)d_in[2];   // [L,128,128]
    const float* b_l   = (const float*)d_in[3];   // [L,128]
    const float* W_r   = (const float*)d_in[4];   // [L,128,128]
    const float* gamma = (const float*)d_in[5];   // [L,128]
    const float* beta  = (const float*)d_in[6];   // [L,128]
    const float* W_o   = (const float*)d_in[7];   // [128,64]
    const float* b_o   = (const float*)d_in[8];   // [64]
    float* out = (float*)d_out;

    const int N = in_sizes[0] / D_HID;
    const int E = in_sizes[1] / 2;
    const int L = in_sizes[2] / (D_HID * D_HID);

    // workspace carve-up (256B aligned)
    char* ws = (char*)d_ws;
    size_t off = 0;
    auto carve = [&](size_t bytes) -> void* {
        void* p = ws + off;
        off = (off + bytes + 255) & ~(size_t)255;
        return p;
    };
    float*     hA    = (float*)carve((size_t)N * D_HID * 4);
    float*     hB    = (float*)carve((size_t)N * D_HID * 4);
    float*     agg   = (float*)carve((size_t)N * D_HID * 4);
    float*     invd  = (float*)carve((size_t)N * 4);
    float*     stats = (float*)carve(256 * 4);               // colsum[128] | colsumsq[128]
    _Float16*  wl16  = (_Float16*)carve(4 * 8 * 32 * 16 * 2);
    _Float16*  wr16  = (_Float16*)carve(4 * 8 * 32 * 16 * 2);
    _Float16*  wo16  = (_Float16*)carve(4 * 4 * 32 * 16 * 2);
    (void)ws_size; (void)n_in; (void)out_size;

    const int TPB = 256;
    // ---- in-degree -> inverse ----
    zero_f32_kernel<<<(N + TPB - 1) / TPB, TPB, 0, stream>>>(invd, N);
    degree_kernel<<<(E + TPB - 1) / TPB, TPB, 0, stream>>>(ei, invd, E);
    invdeg_kernel<<<(N + TPB - 1) / TPB, TPB, 0, stream>>>(invd, N);

    const float* hin = x;
    for (int l = 0; l < L; ++l) {
        float* hout = (l & 1) ? hB : hA;
        zero_f32_kernel<<<((size_t)N * D_HID + TPB - 1) / TPB, TPB, 0, stream>>>(agg, N * D_HID);
        zero_f32_kernel<<<1, 256, 0, stream>>>(stats, 256);
        pack_b16_kernel<<<(4 * 8 * 32 + 127) / 128, 128, 0, stream>>>(
            W_l + (size_t)l * D_HID * D_HID, wl16, 4, 8);
        pack_b16_kernel<<<(4 * 8 * 32 + 127) / 128, 128, 0, stream>>>(
            W_r + (size_t)l * D_HID * D_HID, wr16, 4, 8);
        scatter_kernel<<<((size_t)E * 32 + TPB - 1) / TPB, TPB, 0, stream>>>(hin, ei, agg, E);
        sage_gemm_kernel<<<(N + 15) / 16, 256, 0, stream>>>(
            agg, hin, invd, wl16, wr16, b_l + (size_t)l * D_HID, hout,
            stats, stats + D_HID, N);
        bn_relu_kernel<<<((size_t)N * (D_HID / 4) + TPB - 1) / TPB, TPB, 0, stream>>>(
            hout, stats, stats + D_HID, gamma + (size_t)l * D_HID, beta + (size_t)l * D_HID, N);
        hin = hout;
    }

    pack_b16_kernel<<<(4 * 4 * 32 + 127) / 128, 128, 0, stream>>>(W_o, wo16, 4, 4);
    out_gemm_kernel<<<(N + 15) / 16, 128, 0, stream>>>(hin, wo16, b_o, out, N);
}
